// MODEL_89507118449160
// MI455X (gfx1250) — matrix operand order repaired and checked by reading, not compiled, not run on hardware
//
#include <hip/hip_runtime.h>
#include <hip/hip_bf16.h>
#include <math.h>

// ---------------- problem constants ----------------
constexpr int kBS   = 8;
constexpr int kSL   = 100;
constexpr int kNN   = 200;          // nodes per graph
constexpr int kNE   = 600;          // edges per graph
constexpr int kCP1  = 111;          // C+1
constexpr int kD    = 128;          // GGNN dim
constexpr int kDC   = 64;           // transdim
constexpr int kH    = 256;          // LSTM hidden
constexpr int kFEAT = 2 + kCP1 + kDC;  // 177
constexpr int kG    = kBS * kSL;       // 800
constexpr int kR    = kG * kNN;        // 160000 rows (multiple of 16)

static_assert(kR % 16 == 0, "row tiles");
static_assert(kD % 32 == 0, "k tiles");

// ---------------- WMMA types ----------------
typedef __attribute__((ext_vector_type(16))) __bf16 v16bf;
typedef __attribute__((ext_vector_type(8)))  float  v8f;

__device__ __forceinline__ float sigf(float x) { return 1.f / (1.f + expf(-x)); }

// A fragment: 16x32 fp32 sub-block (row stride kD), converted to bf16.
// Layout (contiguous K, matching B): lane<16: K 0..15; lane>=16: K 16..31.
__device__ __forceinline__ v16bf load_a_frag(const float* __restrict__ tile, int lane) {
  int m = lane & 15;
  const float* rp = tile + (size_t)m * kD;
  int klo = (lane < 16) ? 0 : 16;
  int khi = klo + 8;
  v16bf a;
#pragma unroll
  for (int i = 0; i < 8; ++i) a[i] = (__bf16)rp[klo + i];
#pragma unroll
  for (int i = 0; i < 8; ++i) a[8 + i] = (__bf16)rp[khi + i];
  return a;
}

// B fragment from "transposed weight" wt[n][k] (row stride kD, bf16):
// lane holds col n = n0 + lane%16, K = k0 + 16*(lane/16) .. +15 -> 16 contiguous bf16.
__device__ __forceinline__ v16bf load_b_frag(const __bf16* __restrict__ wt, int n0, int k0, int lane) {
  const __bf16* p = wt + (size_t)(n0 + (lane & 15)) * kD + k0 + ((lane >> 4) << 4);
  v16bf b;
#pragma unroll
  for (int i = 0; i < 16; ++i) b[i] = p[i];
  return b;
}

// ---------------- utility kernels ----------------
__global__ void fill_zero_kernel(float* __restrict__ p, size_t n) {
  size_t i = (size_t)blockIdx.x * blockDim.x + threadIdx.x;
  size_t stride = (size_t)gridDim.x * blockDim.x;
  for (; i < n; i += stride) p[i] = 0.f;
}

__global__ void convert_bf16_kernel(const float* __restrict__ s, __bf16* __restrict__ d, int n) {
  int i = blockIdx.x * blockDim.x + threadIdx.x;
  if (i < n) d[i] = (__bf16)s[i];
}

// ggnn_w: (4, D, D) with w[l][d][e]; store transposed bf16 wt[l][e][d]
__global__ void prep_ggnn_w_kernel(const float* __restrict__ w, __bf16* __restrict__ wt) {
  int i = blockIdx.x * blockDim.x + threadIdx.x;
  if (i >= 4 * kD * kD) return;
  int l = i / (kD * kD);
  int r = i - l * kD * kD;
  int d = r / kD, e = r - d * kD;
  wt[(size_t)l * kD * kD + (size_t)e * kD + d] = (__bf16)w[i];
}

__global__ void typemean_kernel(const float* __restrict__ eew, float* __restrict__ tm) {
  int t = threadIdx.x;
  if (t < 8) {
    float s = 0.f;
    for (int d = 0; d < kD; ++d) s += eew[t * kD + d];
    tm[t] = s / (float)kD;
  }
}

__global__ void embed_kernel(const int* __restrict__ node_id, const float* __restrict__ emb,
                             float* __restrict__ x) {
  size_t i = (size_t)blockIdx.x * blockDim.x + threadIdx.x;
  size_t total = (size_t)kR * kD;
  size_t stride = (size_t)gridDim.x * blockDim.x;
  for (; i < total; i += stride) {
    size_t row = i >> 7;          // / kD
    int d = (int)(i & (kD - 1));
    x[i] = emb[(size_t)node_id[row] * kD + d];
  }
}

// ---------------- WMMA GEMM: out[R x NCOLS] = A[R x 128] * Wt^T ----------------
// One wave owns a full 16-row stripe: each A fragment is loaded once per K-chunk
// and reused across all TN column tiles (register-blocked over N).
template <int NCOLS, bool OUT_BF16>
__global__ void gemm_kernel(const float* __restrict__ A, const __bf16* __restrict__ Wt,
                            float* __restrict__ outf, __bf16* __restrict__ outb) {
  constexpr int TN = NCOLS / 16;
  int tm = blockIdx.x * (blockDim.x >> 5) + (threadIdx.x >> 5);   // stripe (row tile)
  int lane = threadIdx.x & 31;
  const float* ab = A + (size_t)tm * 16 * kD;
  // prefetch A data 8 stripes ahead, this wave's stripe in the next block (speculative; dropped if invalid)
  __builtin_prefetch(ab + (size_t)(256 >> 5) * 16 * kD, 0, 1);
  v8f zero = {};
  v8f acc[TN];
#pragma unroll
  for (int t = 0; t < TN; ++t) acc[t] = zero;
#pragma unroll
  for (int kc = 0; kc < kD / 32; ++kc) {
    v16bf a = load_a_frag(ab + kc * 32, lane);
#pragma unroll
    for (int t = 0; t < TN; ++t) {
      v16bf b = load_b_frag(Wt, t * 16, kc * 32, lane);
      acc[t] = __builtin_amdgcn_wmma_f32_16x16x32_bf16(false, a, false, b, (short)0, acc[t], false, false);
    }
  }
  int nl = lane & 15;
  int rb = tm * 16 + ((lane >> 4) << 3);
#pragma unroll
  for (int t = 0; t < TN; ++t) {
    int n = t * 16 + nl;
#pragma unroll
    for (int j = 0; j < 8; ++j) {
      size_t o = (size_t)(rb + j) * NCOLS + n;
      if (OUT_BF16) outb[o] = (__bf16)acc[t][j];
      else          outf[o] = acc[t][j];
    }
  }
}

// ---------------- edge scatter: agg[g,dst,:] += ew * m[g,src,:] ----------------
__global__ void scatter_kernel(const __bf16* __restrict__ m, const int* __restrict__ edge,
                               const int* __restrict__ etype, const float* __restrict__ tm,
                               float* __restrict__ agg) {
  int ge = blockIdx.x;                 // g*kNE + e
  int g = ge / kNE, e = ge - g * kNE;
  int src = edge[((size_t)g * 2) * kNE + e];
  int dst = edge[((size_t)g * 2 + 1) * kNE + e];
  float w = tm[etype[(size_t)g * kNE + e]];
  int d = threadIdx.x;                 // 128 threads
  float v = w * (float)m[((size_t)g * kNN + src) * kD + d];
  atomicAdd(&agg[((size_t)g * kNN + dst) * kD + d], v);
}

// ---------------- fused GRU: gi=agg@Wih^T+bih, gh=x@Whh^T+bhh, GRU update of x ----
// 8 waves per block; wave w owns 3 gi-tiles and 3 gh-tiles (cols [48w, 48w+48)),
// sharing each agg/x A-fragment across its 3 column tiles.
__global__ void gru_kernel(const float* __restrict__ agg, float* __restrict__ x,
                           const __bf16* __restrict__ wih, const __bf16* __restrict__ whh,
                           const float* __restrict__ bih, const float* __restrict__ bhh) {
  __shared__ float gbuf[2][16][3 * kD];     // gi / gh tiles, 49152 bytes
  int tileM = blockIdx.x;                   // 16 rows per block
  int wid = threadIdx.x >> 5;               // 8 waves
  int lane = threadIdx.x & 31;
  const float* aggb = agg + (size_t)tileM * 16 * kD;
  float* xb = x + (size_t)tileM * 16 * kD;

  v8f zero = {};
  v8f gi_acc[3], gh_acc[3];
#pragma unroll
  for (int j = 0; j < 3; ++j) { gi_acc[j] = zero; gh_acc[j] = zero; }

#pragma unroll
  for (int kc = 0; kc < kD / 32; ++kc) {
    v16bf a_gi = load_a_frag(aggb + kc * 32, lane);
#pragma unroll
    for (int j = 0; j < 3; ++j) {
      v16bf b = load_b_frag(wih, (wid * 3 + j) * 16, kc * 32, lane);
      gi_acc[j] = __builtin_amdgcn_wmma_f32_16x16x32_bf16(false, a_gi, false, b, (short)0, gi_acc[j], false, false);
    }
    v16bf a_gh = load_a_frag(xb + kc * 32, lane);
#pragma unroll
    for (int j = 0; j < 3; ++j) {
      v16bf b = load_b_frag(whh, (wid * 3 + j) * 16, kc * 32, lane);
      gh_acc[j] = __builtin_amdgcn_wmma_f32_16x16x32_bf16(false, a_gh, false, b, (short)0, gh_acc[j], false, false);
    }
  }
  int rb = (lane >> 4) << 3;
#pragma unroll
  for (int j = 0; j < 3; ++j) {
    int n = (wid * 3 + j) * 16 + (lane & 15);
    float bvi = bih[n];
    float bvh = bhh[n];
#pragma unroll
    for (int jj = 0; jj < 8; ++jj) {
      gbuf[0][rb + jj][n] = gi_acc[j][jj] + bvi;
      gbuf[1][rb + jj][n] = gh_acc[j][jj] + bvh;
    }
  }
  __syncthreads();

  for (int idx = threadIdx.x; idx < 16 * kD; idx += blockDim.x) {
    int r = idx >> 7, d = idx & (kD - 1);
    float ir = gbuf[0][r][d], iz = gbuf[0][r][kD + d], in_ = gbuf[0][r][2 * kD + d];
    float hr = gbuf[1][r][d], hz = gbuf[1][r][kD + d], hn = gbuf[1][r][2 * kD + d];
    float rr = sigf(ir + hr);
    float zz = sigf(iz + hz);
    float nn = tanhf(in_ + rr * hn);
    float xo = xb[(size_t)r * kD + d];
    xb[(size_t)r * kD + d] = (1.f - zz) * nn + zz * xo;
  }
}

// ---------------- attention: per (g,c) block ----------------
__global__ void attn_kernel(const float* __restrict__ c_embed, const float* __restrict__ concept,
                            const float* __restrict__ ggnn, float* __restrict__ attnsum) {
  int g = blockIdx.x / kCP1;
  int c = blockIdx.x - g * kCP1;
  float ce = c_embed[(size_t)g * kCP1 + c];
  if (ce == 0.f) return;                              // contribution is exactly zero (block-uniform)
  __shared__ float q[kDC];
  __shared__ float sc[kNN];
  __shared__ float red[256];
  int tid = threadIdx.x;
  if (tid < kDC) q[tid] = ce * concept[(size_t)c * kDC + tid];
  __syncthreads();
  const float* gb = ggnn + (size_t)g * kNN * kDC;
  if (tid < kNN) {
    float s = 0.f;
    for (int d = 0; d < kDC; ++d) s += q[d] * gb[(size_t)tid * kDC + d];
    sc[tid] = s;
  }
  __syncthreads();
  // softmax over n = 0..199
  red[tid] = (tid < kNN) ? sc[tid] : -INFINITY;
  __syncthreads();
  for (int s = 128; s > 0; s >>= 1) { if (tid < s) red[tid] = fmaxf(red[tid], red[tid + s]); __syncthreads(); }
  float mx = red[0];
  __syncthreads();
  float ex = (tid < kNN) ? expf(sc[tid] - mx) : 0.f;
  red[tid] = ex;
  __syncthreads();
  for (int s = 128; s > 0; s >>= 1) { if (tid < s) red[tid] += red[tid + s]; __syncthreads(); }
  float denom = red[0];
  __syncthreads();
  if (tid < kNN) sc[tid] = ex / denom;
  __syncthreads();
  if (tid < kDC) {
    float a = 0.f;
    for (int n = 0; n < kNN; ++n) a += sc[n] * gb[(size_t)n * kDC + tid];
    atomicAdd(&attnsum[(size_t)g * kDC + tid], ce * a);
  }
}

// ---------------- lstm_in = concat(c_embed, attnsum/num, cur_result) ----------------
__global__ void lstmin_kernel(const float* __restrict__ c_id, const float* __restrict__ c_embed,
                              const float* __restrict__ attnsum, const float* __restrict__ cur_result,
                              float* __restrict__ lstmin) {
  __shared__ float red[256];
  int g = blockIdx.x, tid = threadIdx.x;
  float p = 0.f;
  for (int c = tid; c < kCP1; c += 256) p += c_id[(size_t)g * kCP1 + c];
  red[tid] = p;
  __syncthreads();
  for (int s = 128; s > 0; s >>= 1) { if (tid < s) red[tid] += red[tid + s]; __syncthreads(); }
  float num = red[0];
  if (num == 0.f) num = 1.f;
  for (int i = tid; i < kFEAT; i += 256) {
    float v;
    if (i < kCP1)            v = c_embed[(size_t)g * kCP1 + i];
    else if (i < kCP1 + kDC) v = attnsum[(size_t)g * kDC + (i - kCP1)] / num;
    else                     v = cur_result[(size_t)g * 2 + (i - kCP1 - kDC)];
    lstmin[(size_t)g * kFEAT + i] = v;
  }
}

// ---------------- LSTM scan (single block, 1024 threads) ----------------
__global__ void lstm_kernel(const float* __restrict__ lstmin, const float* __restrict__ wih,
                            const float* __restrict__ whh, const float* __restrict__ bih,
                            const float* __restrict__ bhh, float* __restrict__ lstmout) {
  __shared__ float hsh[kBS][kH];
  __shared__ float csh[kBS][kH];
  __shared__ float gates[kBS][4 * kH];
  int tid = threadIdx.x;  // 1024 = 4*kH
  for (int i = tid; i < kBS * kH; i += 1024) { hsh[i / kH][i % kH] = 0.f; csh[i / kH][i % kH] = 0.f; }
  __syncthreads();
  const float* wi = wih + (size_t)tid * kFEAT;
  const float* wh = whh + (size_t)tid * kH;
  float bb = bih[tid] + bhh[tid];
  for (int t = 0; t < kSL; ++t) {
    for (int b = 0; b < kBS; ++b) {
      const float* xt = lstmin + ((size_t)b * kSL + t) * kFEAT;
      float acc = bb;
      for (int k = 0; k < kFEAT; ++k) acc += xt[k] * wi[k];
      for (int k = 0; k < kH; ++k) acc += hsh[b][k] * wh[k];
      gates[b][tid] = acc;
    }
    __syncthreads();
#pragma unroll
    for (int rep = 0; rep < 2; ++rep) {
      int idx = tid + rep * 1024;
      int b = idx >> 8, hh = idx & (kH - 1);
      float ig = gates[b][hh];
      float fg = gates[b][kH + hh];
      float gg = gates[b][2 * kH + hh];
      float og = gates[b][3 * kH + hh];
      float cn = sigf(fg) * csh[b][hh] + sigf(ig) * tanhf(gg);
      float hn = sigf(og) * tanhf(cn);
      csh[b][hh] = cn;
      hsh[b][hh] = hn;
      lstmout[((size_t)b * kSL + t) * kH + hh] = hn;
    }
    __syncthreads();
  }
}

// ---------------- prediction + BCE ----------------
__global__ void predloss_kernel(const float* __restrict__ lstmout, const float* __restrict__ pred_w,
                                const float* __restrict__ pred_b, const float* __restrict__ target_c,
                                const float* __restrict__ result, float* __restrict__ out,
                                float* __restrict__ lossacc) {
  __shared__ float red1[256], red2[256];
  int g = blockIdx.x, tid = threadIdx.x;
  const float* lo = lstmout + (size_t)g * kH;
  const float* tc = target_c + (size_t)g * kCP1;
  float p1 = 0.f, ncs = 0.f;
  for (int c = tid; c < kCP1; c += 256) {
    float tcv = tc[c];
    ncs += tcv;
    if (tcv != 0.f) {
      float dp = pred_b[c];
      const float* pw = pred_w + (size_t)c * kH;
      for (int k = 0; k < kH; ++k) dp += lo[k] * pw[k];
      p1 += tcv * dp;
    }
  }
  red1[tid] = p1; red2[tid] = ncs;
  __syncthreads();
  for (int s = 128; s > 0; s >>= 1) {
    if (tid < s) { red1[tid] += red1[tid + s]; red2[tid] += red2[tid + s]; }
    __syncthreads();
  }
  if (tid == 0) {
    float nc = red2[0];
    bool mask = nc > 0.f;
    float fp = red1[0] / (mask ? nc : 1.f);
    float ft = result[g];
    float bce = fmaxf(fp, 0.f) - fp * ft + log1pf(expf(-fabsf(fp)));
    float mf = mask ? 1.f : 0.f;
    atomicAdd(&lossacc[0], bce * mf);
    atomicAdd(&lossacc[1], mf);
    out[1 + g] = sigf(fp);
    out[1 + kG + g] = ft;
  }
}

__global__ void finalize_kernel(const float* __restrict__ lossacc, float* __restrict__ out) {
  out[0] = lossacc[0] / fmaxf(lossacc[1], 1.f);
}

// ---------------- host ----------------
extern "C" void kernel_launch(void* const* d_in, const int* in_sizes, int n_in,
                              void* d_out, int out_size, void* d_ws, size_t ws_size,
                              hipStream_t stream) {
  // inputs (setup_inputs dict order)
  const float* c_id        = (const float*)d_in[1];
  const int*   node_id     = (const int*)d_in[2];
  const int*   edge        = (const int*)d_in[3];
  const int*   edge_type   = (const int*)d_in[4];
  const float* target_c    = (const float*)d_in[5];
  const float* result      = (const float*)d_in[6];
  const float* c_embed     = (const float*)d_in[7];
  const float* cur_result  = (const float*)d_in[8];
  const float* node_emb_w  = (const float*)d_in[9];
  const float* edge_emb_w  = (const float*)d_in[10];
  const float* ggnn_w      = (const float*)d_in[11];
  const float* gru_w_ih    = (const float*)d_in[12];
  const float* gru_w_hh    = (const float*)d_in[13];
  const float* gru_b_ih    = (const float*)d_in[14];
  const float* gru_b_hh    = (const float*)d_in[15];
  const float* transdim_w  = (const float*)d_in[16];
  const float* concept_emb = (const float*)d_in[17];
  const float* lstm_w_ih   = (const float*)d_in[18];
  const float* lstm_w_hh   = (const float*)d_in[19];
  const float* lstm_b_ih   = (const float*)d_in[20];
  const float* lstm_b_hh   = (const float*)d_in[21];
  const float* pred_w      = (const float*)d_in[22];
  const float* pred_b      = (const float*)d_in[23];
  float* out = (float*)d_out;

  // workspace bump allocator (256B aligned)
  char* ws = (char*)d_ws;
  size_t off = 0;
  auto alloc = [&](size_t bytes) -> void* {
    off = (off + 255) & ~(size_t)255;
    void* p = ws + off;
    off += bytes;
    return p;
  };
  float*  x       = (float*) alloc((size_t)kR * kD * 4);
  float*  agg     = (float*) alloc((size_t)kR * kD * 4);
  __bf16* mb      = (__bf16*)alloc((size_t)kR * kD * 2);
  float*  ggnnb   = (float*) alloc((size_t)kR * kDC * 4);
  __bf16* wgt     = (__bf16*)alloc((size_t)4 * kD * kD * 2);
  __bf16* wihb    = (__bf16*)alloc((size_t)3 * kD * kD * 2);
  __bf16* whhb    = (__bf16*)alloc((size_t)3 * kD * kD * 2);
  __bf16* wtdb    = (__bf16*)alloc((size_t)kDC * kD * 2);
  float*  tmean   = (float*) alloc(8 * 4);
  float*  attnsum = (float*) alloc((size_t)kG * kDC * 4);
  float*  lstmin  = (float*) alloc((size_t)kG * kFEAT * 4);
  float*  lstmout = (float*) alloc((size_t)kG * kH * 4);
  float*  lossacc = (float*) alloc(2 * 4);
  (void)ws_size; (void)in_sizes; (void)n_in; (void)out_size;

  // --- weight prep ---
  convert_bf16_kernel<<<(3 * kD * kD + 255) / 256, 256, 0, stream>>>(gru_w_ih, wihb, 3 * kD * kD);
  convert_bf16_kernel<<<(3 * kD * kD + 255) / 256, 256, 0, stream>>>(gru_w_hh, whhb, 3 * kD * kD);
  convert_bf16_kernel<<<(kDC * kD + 255) / 256, 256, 0, stream>>>(transdim_w, wtdb, kDC * kD);
  prep_ggnn_w_kernel<<<(4 * kD * kD + 255) / 256, 256, 0, stream>>>(ggnn_w, wgt);
  typemean_kernel<<<1, 256, 0, stream>>>(edge_emb_w, tmean);

  // --- embedding gather ---
  embed_kernel<<<20000, 256, 0, stream>>>(node_id, node_emb_w, x);

  // --- 4 GGNN layers ---
  for (int l = 0; l < 4; ++l) {
    gemm_kernel<kD, true><<<(kR / 16) / 8, 256, 0, stream>>>(
        x, wgt + (size_t)l * kD * kD, nullptr, mb);
    fill_zero_kernel<<<4096, 256, 0, stream>>>(agg, (size_t)kR * kD);
    scatter_kernel<<<kG * kNE, kD, 0, stream>>>(mb, edge, edge_type, tmean, agg);
    gru_kernel<<<kR / 16, 256, 0, stream>>>(agg, x, wihb, whhb, gru_b_ih, gru_b_hh);
  }

  // --- transdim projection ---
  gemm_kernel<kDC, false><<<(kR / 16) / 8, 256, 0, stream>>>(x, wtdb, ggnnb, nullptr);

  // --- attention ---
  fill_zero_kernel<<<256, 256, 0, stream>>>(attnsum, (size_t)kG * kDC);
  fill_zero_kernel<<<1, 64, 0, stream>>>(lossacc, 2);
  attn_kernel<<<kG * kCP1, 256, 0, stream>>>(c_embed, concept_emb, ggnnb, attnsum);

  // --- LSTM input, scan, prediction, loss ---
  lstmin_kernel<<<kG, 256, 0, stream>>>(c_id, c_embed, attnsum, cur_result, lstmin);
  lstm_kernel<<<1, 1024, 0, stream>>>(lstmin, lstm_w_ih, lstm_w_hh, lstm_b_ih, lstm_b_hh, lstmout);
  predloss_kernel<<<kG, 256, 0, stream>>>(lstmout, pred_w, pred_b, target_c, result, out, lossacc);
  finalize_kernel<<<1, 1, 0, stream>>>(lossacc, out);
}
